// TaylorModel_25280177504598
// MI455X (gfx1250) — matrix operand order repaired and checked by reading, not compiled, not run on hardware
//
#include <hip/hip_runtime.h>
#include <hip/hip_bf16.h>

typedef __attribute__((ext_vector_type(16))) _Float16 v16h;
typedef __attribute__((ext_vector_type(8)))  float    v8f;

#define WAVES 2            // waves per block (wave32) -> 64 rows per block
#define ROWS_PER_WAVE 32   // two 16-row M tiles per wave (B fragments reused 2x)
#define KPAD2 256          // layer-2 K padded (230 -> 256), also LDS row stride
#define L1_CHUNKS 6        // 189 -> 192 = 6 * 32
#define L2_CHUNKS 8        // 230 -> 256 = 8 * 32
#define N_FRAGS (2*L1_CHUNKS + 2*L2_CHUNKS)  // 28 B-matrix fragments

__device__ __forceinline__ float fast_tanh(float x) {
#if __has_builtin(__builtin_amdgcn_tanhf)
    return __builtin_amdgcn_tanhf(x);   // v_tanh_f32 (gfx1250 TRANS op)
#else
    return tanhf(x);
#endif
}

// decode flat upper-triangular index (np.triu_indices order: j outer, k=j..c-1)
__device__ __forceinline__ void decode_pair(int idx, int c, int& j, int& k) {
    int jj = 0, rem = idx;
    while (rem >= c - jj) { rem -= (c - jj); ++jj; }
    j = jj; k = jj + rem;
}

union AFrag { v16h v; uint4 q[2]; };

__device__ __forceinline__ v8f wmma_f16(const v16h& a, const v16h& b, const v8f& c) {
    return __builtin_amdgcn_wmma_f32_16x16x32_f16(false, a, false, b, (short)0, c,
                                                  false, false);
}

// ---------------------------------------------------------------------------
// Prep kernel: pack W1 (20x189) and W2 (20x230) into per-lane WMMA B fragments.
// B (32x16, f16) K order chosen here (same as A): lanes 0-15 : VGPR v holds K=2v,2v+1 at N=lane
//                        lanes 16-31: VGPR v holds K=16+2v,16+2v+1 at N=lane-16
// => lane's 16 halves are K = kb..kb+15 with kb = chunk*32 + (lane<16 ? 0 : 16).
// Fragment f stored at frags[f*512 + lane*16 + e].
// ---------------------------------------------------------------------------
__global__ void build_b_frags(const float* __restrict__ W1,
                              const float* __restrict__ W2,
                              _Float16* __restrict__ frags) {
    const int b    = blockIdx.x;      // fragment index 0..27
    const int lane = threadIdx.x;     // 0..31
    const float* W; int Kdim, chunk, ntile;
    if (b < 2 * L1_CHUNKS) { W = W1; Kdim = 189; chunk = b >> 1;                 ntile = b & 1; }
    else                   { W = W2; Kdim = 230; chunk = (b - 2*L1_CHUNKS) >> 1; ntile = b & 1; }
    const int n  = ntile * 16 + (lane & 15);
    const int kb = chunk * 32 + ((lane < 16) ? 0 : 16);
    v16h vals;
#pragma unroll
    for (int e = 0; e < 16; ++e) {
        const int K = kb + e;
        const float v = (n < 20 && K < Kdim) ? W[n * Kdim + K] : 0.0f;
        vals[e] = (_Float16)v;
    }
    *(v16h*)(frags + (size_t)b * 512 + lane * 16) = vals;   // 32B coalesced
}

// ---------------------------------------------------------------------------
// Main kernel: one wave = 32 rows (two 16-row M tiles sharing each B fragment).
// ---------------------------------------------------------------------------
__global__ __launch_bounds__(32 * WAVES)
void taylor_mlp_kernel(const float* __restrict__ x,
                       const float* __restrict__ b1,
                       const float* __restrict__ b2,
                       const float* __restrict__ W3,
                       const float* __restrict__ b3,
                       const _Float16* __restrict__ frags,
                       float* __restrict__ out, int nRows) {
    __shared__ __align__(16) float    s_xh [WAVES][ROWS_PER_WAVE * 20];    // x / h1 / h2
    __shared__ __align__(16) _Float16 s_aug[WAVES][ROWS_PER_WAVE * KPAD2]; // augmented feats

    const int w    = threadIdx.x >> 5;
    const int lane = threadIdx.x & 31;
    const int row0 = (blockIdx.x * WAVES + w) * ROWS_PER_WAVE;

    float*    xh  = s_xh[w];
    _Float16* aug = s_aug[w];
    const v16h* fragv = (const v16h*)frags;

    // ---- load x tile [32 x 18] into LDS (zero-pad tail rows) ----
    for (int i = lane; i < ROWS_PER_WAVE * 18; i += 32) {
        const int r = i / 18, c = i % 18;
        const int gr = row0 + r;
        xh[r * 20 + c] = (gr < nRows) ? x[(size_t)gr * 18 + c] : 0.0f;
    }

    // ---- augment layer-1 input: 18 linear + 171 products -> 192 (padded) ----
    for (int f = lane; f < L1_CHUNKS * 32; f += 32) {
        if (f < 18) {
#pragma unroll
            for (int r = 0; r < ROWS_PER_WAVE; ++r)
                aug[r * KPAD2 + f] = (_Float16)xh[r * 20 + f];
        } else if (f < 189) {
            int j, k; decode_pair(f - 18, 18, j, k);
#pragma unroll
            for (int r = 0; r < ROWS_PER_WAVE; ++r)
                aug[r * KPAD2 + f] = (_Float16)(xh[r * 20 + j] * xh[r * 20 + k]);
        } else {
#pragma unroll
            for (int r = 0; r < ROWS_PER_WAVE; ++r)
                aug[r * KPAD2 + f] = (_Float16)0.0f;
        }
    }

    // A fragment (16-bit 16x32), K order chosen here (same as B): lanes 0-15 hold K 0-15, lanes 16-31 hold
    // K 16-31 -> two contiguous 16B groups per lane per chunk.
    const _Float16* augRow0 = aug + (lane & 15) * KPAD2;            // M tile 0 (rows 0-15)
    const _Float16* augRow1 = augRow0 + 16 * KPAD2;                 // M tile 1 (rows 16-31)
    const int off0 = (lane < 16) ? 0 : 16;  // first K-group offset (halves)

    // ---- layer 1: 6 K-chunks x 2 M-tiles x 2 N-tiles of v_wmma_f32_16x16x32_f16 ----
    v8f a00 = {}, a01 = {}, a10 = {}, a11 = {};   // [mtile][ntile]
    for (int ch = 0; ch < L1_CHUNKS; ++ch) {
        AFrag af0, af1;
        const _Float16* p0 = augRow0 + ch * 32 + off0;
        const _Float16* p1 = augRow1 + ch * 32 + off0;
        af0.q[0] = *(const uint4*)p0;         af0.q[1] = *(const uint4*)(p0 + 8);
        af1.q[0] = *(const uint4*)p1;         af1.q[1] = *(const uint4*)(p1 + 8);
        const v16h bv0 = fragv[(size_t)(ch * 2 + 0) * 32 + lane];
        const v16h bv1 = fragv[(size_t)(ch * 2 + 1) * 32 + lane];
        a00 = wmma_f16(af0.v, bv0, a00);
        a01 = wmma_f16(af0.v, bv1, a01);
        a10 = wmma_f16(af1.v, bv0, a10);
        a11 = wmma_f16(af1.v, bv1, a11);
    }

    // ---- bias + tanh, scatter h1 [32 x 20] back to LDS ----
    {
        const int col = lane & 15;
        const float bA = b1[col];
        const float bB = (col < 4) ? b1[16 + col] : 0.0f;
        const int rbase = (lane >> 4) << 3;   // C/D layout: lanes>=16 hold M=8..15
#pragma unroll
        for (int r = 0; r < 8; ++r) {
            const int rowA = rbase + r;       // M tile 0
            const int rowB = 16 + rowA;       // M tile 1
            xh[rowA * 20 + col] = fast_tanh(a00[r] + bA);
            xh[rowB * 20 + col] = fast_tanh(a10[r] + bA);
            if (col < 4) {
                xh[rowA * 20 + 16 + col] = fast_tanh(a01[r] + bB);
                xh[rowB * 20 + 16 + col] = fast_tanh(a11[r] + bB);
            }
        }
    }

    // ---- augment layer-2 input: 20 linear + 210 products -> 256 (padded) ----
    for (int f = lane; f < KPAD2; f += 32) {
        if (f < 20) {
#pragma unroll
            for (int r = 0; r < ROWS_PER_WAVE; ++r)
                aug[r * KPAD2 + f] = (_Float16)xh[r * 20 + f];
        } else if (f < 230) {
            int j, k; decode_pair(f - 20, 20, j, k);
#pragma unroll
            for (int r = 0; r < ROWS_PER_WAVE; ++r)
                aug[r * KPAD2 + f] = (_Float16)(xh[r * 20 + j] * xh[r * 20 + k]);
        } else {
#pragma unroll
            for (int r = 0; r < ROWS_PER_WAVE; ++r)
                aug[r * KPAD2 + f] = (_Float16)0.0f;
        }
    }

    // ---- layer 2: 8 K-chunks x 2 M-tiles x 2 N-tiles ----
    v8f d00 = {}, d01 = {}, d10 = {}, d11 = {};
    for (int ch = 0; ch < L2_CHUNKS; ++ch) {
        AFrag af0, af1;
        const _Float16* p0 = augRow0 + ch * 32 + off0;
        const _Float16* p1 = augRow1 + ch * 32 + off0;
        af0.q[0] = *(const uint4*)p0;         af0.q[1] = *(const uint4*)(p0 + 8);
        af1.q[0] = *(const uint4*)p1;         af1.q[1] = *(const uint4*)(p1 + 8);
        const v16h bv0 = fragv[(size_t)(2*L1_CHUNKS + ch * 2 + 0) * 32 + lane];
        const v16h bv1 = fragv[(size_t)(2*L1_CHUNKS + ch * 2 + 1) * 32 + lane];
        d00 = wmma_f16(af0.v, bv0, d00);
        d01 = wmma_f16(af0.v, bv1, d01);
        d10 = wmma_f16(af1.v, bv0, d10);
        d11 = wmma_f16(af1.v, bv1, d11);
    }

    // ---- bias + tanh, scatter h2 [32 x 20] back to LDS ----
    {
        const int col = lane & 15;
        const float bA = b2[col];
        const float bB = (col < 4) ? b2[16 + col] : 0.0f;
        const int rbase = (lane >> 4) << 3;
#pragma unroll
        for (int r = 0; r < 8; ++r) {
            const int rowA = rbase + r;
            const int rowB = 16 + rowA;
            xh[rowA * 20 + col] = fast_tanh(d00[r] + bA);
            xh[rowB * 20 + col] = fast_tanh(d10[r] + bA);
            if (col < 4) {
                xh[rowA * 20 + 16 + col] = fast_tanh(d01[r] + bB);
                xh[rowB * 20 + 16 + col] = fast_tanh(d11[r] + bB);
            }
        }
    }

    // ---- layer 3: y = h2 . W3 + b3, one row per lane (all 32 lanes) ----
    {
        const int gr = row0 + lane;
        if (gr < nRows) {
            float acc = b3[0];
#pragma unroll
            for (int n = 0; n < 20; ++n)
                acc = fmaf(xh[lane * 20 + n], W3[n], acc);
            out[gr] = acc;
        }
    }
}

extern "C" void kernel_launch(void* const* d_in, const int* in_sizes, int n_in,
                              void* d_out, int out_size, void* d_ws, size_t ws_size,
                              hipStream_t stream) {
    const float* x  = (const float*)d_in[0];
    const float* W1 = (const float*)d_in[1];
    const float* b1 = (const float*)d_in[2];
    const float* W2 = (const float*)d_in[3];
    const float* b2 = (const float*)d_in[4];
    const float* W3 = (const float*)d_in[5];
    const float* b3 = (const float*)d_in[6];
    float* out = (float*)d_out;

    _Float16* frags = (_Float16*)d_ws;   // 28 frags * 1 KiB = 28 KiB scratch

    const int nRows = in_sizes[0] / 18;  // 262144

    build_b_frags<<<N_FRAGS, 32, 0, stream>>>(W1, W2, frags);

    const int rowsPerBlock = WAVES * ROWS_PER_WAVE;  // 64
    const int blocks = (nRows + rowsPerBlock - 1) / rowsPerBlock;
    taylor_mlp_kernel<<<blocks, 32 * WAVES, 0, stream>>>(x, b1, b2, W3, b3,
                                                         frags, out, nRows);
}
